// QuantizedLinearInt8_74783970558154
// MI455X (gfx1250) — matrix operand order repaired and checked by reading, not compiled, not run on hardware
//
#include <hip/hip_runtime.h>

// ---------------------------------------------------------------------------
// Quantized int8-weight linear (w8a16 style) for MI455X / gfx1250.
//   y[m,n] = sum_k x[m,k] * (q[n,k] * inv_scale[n]) + bias[n]
// q integer in [-127,127] -> exact in bf16; x -> bf16 via v_cvt_pk_bf16_f32;
// v_wmma_f32_16x16x32_bf16 with f32 accum; per-channel scale+bias epilogue.
//
// Pipeline: ping-pong LDS tiles, 1 barrier per k-tile. Fill uses grouped
// register staging (8-chunk load clauses) so global loads batch; compute
// phase keeps 4 B frags resident and double-buffers the A frag so each
// 4-WMMA group hides the next group's LDS latency (loose dscnt waits).
// WG tile 256x128x64, 8 waves 4(M)x2(N), wave tile 64x64 (16 accums).
// ---------------------------------------------------------------------------

typedef __attribute__((ext_vector_type(16))) __bf16        v16bf;
typedef __attribute__((ext_vector_type(4)))  __bf16        v4bf;
typedef __attribute__((ext_vector_type(8)))  float         v8f;
typedef __attribute__((ext_vector_type(4)))  float         v4f;
typedef __attribute__((ext_vector_type(4)))  int           v4i;
typedef __attribute__((ext_vector_type(4)))  unsigned int  v4u;

#define M_TOTAL 8192   // B*T = 4*2048
#define N_TOTAL 4096   // OUT
#define K_TOTAL 4096   // IN

#define BM 256         // workgroup M tile
#define BN 128         // workgroup N tile
#define BK 64          // K tile (two WMMA k-steps of 32)
#define NT (K_TOTAL / BK)   // 64 k-tiles
#define AST 72         // LDS row stride (bf16 elems): 144B, 16B-aligned, padded
#define BST 72

union FragU { v4u u[2]; v16bf bf; };   // 32B: two ds_load_b128 -> one WMMA operand

__device__ __forceinline__ v4bf cvt4f(v4f a) {
  v4bf t = { (__bf16)a.x, (__bf16)a.y, (__bf16)a.z, (__bf16)a.w };
  return t;
}
__device__ __forceinline__ v4bf cvt4i(v4i b) {
  v4bf t = { (__bf16)(float)b.x, (__bf16)(float)b.y,
             (__bf16)(float)b.z, (__bf16)(float)b.w };
  return t;
}

// Load one k-tile from global, convert to bf16, store into LDS buffers.
// Grouped staging: 8-load clauses in distinct registers; next group's loads
// overlap previous group's cvt+store. Peak staging ~64 VGPRs.
__device__ __forceinline__ void fill_tile(const float* __restrict__ pA,
                                          const int*   __restrict__ pB,
                                          int kt,
                                          unsigned short* As_, unsigned short* Bs_,
                                          int sOff) {
  v4f a0[8], a1[8];
  v4i b0[8];
#pragma unroll
  for (int i = 0; i < 8; ++i)
    a0[i] = *(const v4f*)(pA + kt + (size_t)i * 16 * K_TOTAL);
#pragma unroll
  for (int i = 0; i < 8; ++i)
    a1[i] = *(const v4f*)(pA + kt + (size_t)(i + 8) * 16 * K_TOTAL);
#pragma unroll
  for (int i = 0; i < 8; ++i)
    *(v4bf*)&As_[sOff + i * 16 * AST] = cvt4f(a0[i]);
#pragma unroll
  for (int i = 0; i < 8; ++i)
    b0[i] = *(const v4i*)(pB + kt + (size_t)i * 16 * K_TOTAL);
#pragma unroll
  for (int i = 0; i < 8; ++i)
    *(v4bf*)&As_[sOff + (i + 8) * 16 * AST] = cvt4f(a1[i]);
#pragma unroll
  for (int i = 0; i < 8; ++i)
    *(v4bf*)&Bs_[sOff + i * 16 * BST] = cvt4i(b0[i]);
}

__device__ __forceinline__ void compute_tile(const unsigned short* As_,
                                             const unsigned short* Bs_,
                                             int aOff, int bOff, v8f acc[4][4]) {
#pragma unroll
  for (int ks = 0; ks < 2; ++ks) {
    const int ko = ks * 32;
    // B frags resident (4 x 8 VGPRs), ISA 16-bit B 32x16 layout:
    //   lane = column n; lane<16: K = ko+{0..7} then ko+{16..23}; lane>=16: +8 on both
    FragU bfrag[4];
#pragma unroll
    for (int ni = 0; ni < 4; ++ni) {
      const unsigned short* p = &Bs_[bOff + ni * 16 * BST + ko];
      bfrag[ni].u[0] = *(const v4u*)p;
      bfrag[ni].u[1] = *(const v4u*)(p + 16);
    }
    // A frag double-buffered (2 x 8 VGPRs), ISA 16-bit A 16x32 layout:
    //   lane<16: K = ko+{0..7} then ko+{16..23}; lane>=16: +8 on both
    // Loads for group mi+1 are issued before the WMMAs of group mi, so the
    // WMMA chain hides the LDS latency of the next fragment.
    FragU af[2];
    {
      const unsigned short* p = &As_[aOff + ko];
      af[0].u[0] = *(const v4u*)p;
      af[0].u[1] = *(const v4u*)(p + 16);
    }
#pragma unroll
    for (int mi = 0; mi < 4; ++mi) {
      if (mi < 3) {
        const unsigned short* p = &As_[aOff + (mi + 1) * 16 * AST + ko];
        af[(mi + 1) & 1].u[0] = *(const v4u*)p;
        af[(mi + 1) & 1].u[1] = *(const v4u*)(p + 16);
      }
#pragma unroll
      for (int ni = 0; ni < 4; ++ni)
        acc[mi][ni] = __builtin_amdgcn_wmma_f32_16x16x32_bf16(
            false, af[mi & 1].bf, false, bfrag[ni].bf,
            (short)0, acc[mi][ni], false, false);
    }
  }
}

extern "C" __global__ __launch_bounds__(256)
void qlinear_wmma_bf16(const float* __restrict__ x,
                       const int*   __restrict__ q,
                       const float* __restrict__ scale,
                       const float* __restrict__ bias,
                       float*       __restrict__ out) {
  __shared__ unsigned short As0[BM * AST];   // ping-pong x tiles (bf16)
  __shared__ unsigned short As1[BM * AST];
  __shared__ unsigned short Bs0[BN * BST];   // ping-pong q tiles (bf16)
  __shared__ unsigned short Bs1[BN * BST];

  const int tid   = threadIdx.x;
  const int lane  = tid & 31;
  const int wave  = tid >> 5;
  const int l     = lane & 15;
  const int half  = lane >> 4;
  const int waveM = wave & 3;      // 4 waves along M (64 rows each)
  const int waveN = wave >> 2;     // 2 waves along N (64 cols each)

  const int m0 = blockIdx.y * BM;
  const int n0 = blockIdx.x * BN;

  // Per-thread bases: all further addressing is immediate offsets.
  const int ldRow = tid >> 4;              // 0..15
  const int ldCol = (tid & 15) << 2;       // 0..60, step 4
  const float* pA = x + (size_t)(m0 + ldRow) * K_TOTAL + ldCol;
  const int*   pB = q + (size_t)(n0 + ldRow) * K_TOTAL + ldCol;
  const int sOff  = ldRow * AST + ldCol;   // same stride for A and B tiles

  const int aOff = (waveM * 64 + l) * AST + half * 8;
  const int bOff = (waveN * 64 + l) * BST + half * 8;

  const v8f vzero = {0.f, 0.f, 0.f, 0.f, 0.f, 0.f, 0.f, 0.f};
  v8f acc[4][4];
#pragma unroll
  for (int mi = 0; mi < 4; ++mi)
#pragma unroll
    for (int ni = 0; ni < 4; ++ni) acc[mi][ni] = vzero;

  // ---- prologue: tile 0 -> buffer 0 ---------------------------------------
  fill_tile(pA, pB, 0, As0, Bs0, sOff);
  __syncthreads();

  // ---- steady state: 2 k-tiles per iteration, 1 barrier per k-tile --------
  // Filling buf^1 needs no barrier before compute: buf^1 was last READ in the
  // previous iteration, already separated from these writes by that barrier.
  for (int t = 0; t < NT / 2 - 1; ++t) {
    const int kt = 2 * t;
    fill_tile(pA, pB, (kt + 1) * BK, As1, Bs1, sOff);   // next tile -> buf1
    compute_tile(As0, Bs0, aOff, bOff, acc);            // consume buf0
    __syncthreads();

    fill_tile(pA, pB, (kt + 2) * BK, As0, Bs0, sOff);   // next tile -> buf0
    compute_tile(As1, Bs1, aOff, bOff, acc);            // consume buf1
    __syncthreads();
  }

  // ---- pipeline epilogue: tiles NT-2 (buf0) and NT-1 (buf1) ---------------
  fill_tile(pA, pB, (NT - 1) * BK, As1, Bs1, sOff);
  compute_tile(As0, Bs0, aOff, bOff, acc);
  __syncthreads();
  compute_tile(As1, Bs1, aOff, bOff, acc);

  // ---- output epilogue: per-channel dequant scale + bias ------------------
  // C/D layout: VGPR r holds M = r + 8*half (within the 16x16 tile), N = l.
#pragma unroll
  for (int ni = 0; ni < 4; ++ni) {
    const int n = n0 + waveN * 64 + ni * 16 + l;
    const float sc = scale[n];
    const float bs = bias[n];
#pragma unroll
    for (int mi = 0; mi < 4; ++mi) {
      const int mb = m0 + waveM * 64 + mi * 16 + half * 8;
#pragma unroll
      for (int rr = 0; rr < 8; ++rr) {
        out[(size_t)(mb + rr) * N_TOTAL + n] = acc[mi][ni][rr] * sc + bs;
      }
    }
  }
}

extern "C" void kernel_launch(void* const* d_in, const int* in_sizes, int n_in,
                              void* d_out, int out_size, void* d_ws, size_t ws_size,
                              hipStream_t stream) {
  const float* x  = (const float*)d_in[0];
  const int*   q  = (const int*)d_in[1];
  const float* sc = (const float*)d_in[2];
  const float* bs = (const float*)d_in[3];
  float* out = (float*)d_out;

  dim3 grid(N_TOTAL / BN, M_TOTAL / BM);   // (32, 32)
  dim3 block(256);                          // 8 wave32s, 4(M) x 2(N) wave grid
  hipLaunchKernelGGL(qlinear_wmma_bf16, grid, block, 0, stream, x, q, sc, bs, out);
}
